// GRU_84920093376824
// MI455X (gfx1250) — matrix operand order repaired and checked by reading, not compiled, not run on hardware
//
#include <hip/hip_runtime.h>
#include <math.h>

typedef __bf16 bf16;
typedef __attribute__((ext_vector_type(16))) __bf16    v16bf;
typedef __attribute__((ext_vector_type(8)))  float     v8f;
typedef __attribute__((ext_vector_type(8)))  unsigned  v8u;

#define BB    8
#define SS    256
#define DD    300
#define HH    512
#define ROWS  2048      // B*S
#define KP0   608       // 2*D padded to multiple of 32
#define G3H   1536      // 3*H
#define NGL   50000
#define NSE   25000
#define SCAN_WGS 32

// ---------------- WMMA helpers -------------------------------------------
__device__ __forceinline__ v8f wmma_bf16(v16bf a, v16bf b, v8f c) {
  return __builtin_amdgcn_wmma_f32_16x16x32_bf16(false, a, false, b, (short)0, c, false, false);
}

// A fragment: 16x32 bf16 tile, row-major src (stride in elements, mult of 8).
// lane m=lane&15 row; lanes<16 hold K[0..7]+K[16..23], lanes>=16 K[8..15]+K[24..31]
__device__ __forceinline__ v16bf load_afrag(const bf16* __restrict__ src, int stride) {
  const int lane = threadIdx.x & 31;
  const int m = lane & 15;
  const int hi = lane >> 4;
  const bf16* p = src + m * stride + hi * 8;
  uint4 a0 = *(const uint4*)(p);
  uint4 a1 = *(const uint4*)(p + 16);
  v8u t;
  t[0]=a0.x; t[1]=a0.y; t[2]=a0.z; t[3]=a0.w;
  t[4]=a1.x; t[5]=a1.y; t[6]=a1.z; t[7]=a1.w;
  return __builtin_bit_cast(v16bf, t);
}

// Same but rows >= 8 are zero (batch = 8 < 16)
__device__ __forceinline__ v16bf load_afrag_m8(const bf16* __restrict__ src, int stride) {
  const int lane = threadIdx.x & 31;
  const int m = lane & 15;
  const int hi = lane >> 4;
  v8u t = {0u,0u,0u,0u,0u,0u,0u,0u};
  if (m < 8) {
    const bf16* p = src + m * stride + hi * 8;
    uint4 a0 = *(const uint4*)(p);
    uint4 a1 = *(const uint4*)(p + 16);
    t[0]=a0.x; t[1]=a0.y; t[2]=a0.z; t[3]=a0.w;
    t[4]=a1.x; t[5]=a1.y; t[6]=a1.z; t[7]=a1.w;
  }
  return __builtin_bit_cast(v16bf, t);
}

// B fragment: K x 16 from N-major weights w[n][k] (n = output col = WMMA N).
// lane n=lane&15; lanes<16 hold K[0..7]+K[16..23], lanes>=16 K[8..15]+K[24..31]
__device__ __forceinline__ v16bf load_bfrag(const bf16* __restrict__ w, int stride) {
  const int lane = threadIdx.x & 31;
  const int n = lane & 15;
  const int hi = lane >> 4;
  const bf16* p = w + n * stride + hi * 8;
  uint4 b0 = *(const uint4*)(p);
  uint4 b1 = *(const uint4*)(p + 16);
  v8u t;
  t[0]=b0.x; t[1]=b0.y; t[2]=b0.z; t[3]=b0.w;
  t[4]=b1.x; t[5]=b1.y; t[6]=b1.z; t[7]=b1.w;
  return __builtin_bit_cast(v16bf, t);
}

// ---------------- kernels ------------------------------------------------

__global__ void k_init(unsigned* __restrict__ ctr, bf16* __restrict__ hb0, bf16* __restrict__ hb1) {
  int i = blockIdx.x * blockDim.x + threadIdx.x;
  if (i < 512) ctr[i] = 0u;
  if (i < 2048) { ((unsigned*)hb0)[i] = 0u; ((unsigned*)hb1)[i] = 0u; }
}

// build x0 bf16 [2048, 608], row = s*8+b
__global__ void k_embed(const int* __restrict__ widx, const int* __restrict__ sidx,
                        const float* __restrict__ X, bf16* __restrict__ x0) {
  int row = blockIdx.x;
  int s = row >> 3, b = row & 7;
  int wi = widx[b * SS + s];
  int si = sidx[b * SS + s];
  bf16* dst = x0 + (size_t)row * KP0;
  for (int c = threadIdx.x; c < KP0; c += blockDim.x) {
    float v;
    if (c < DD)          v = X[(size_t)wi * DD + c];
    else if (c < 2 * DD) v = (si == 0) ? 0.f : X[(size_t)si * DD + (c - DD)];
    else                 v = 0.f;
    dst[c] = (bf16)v;
  }
}

// fp32 -> bf16 with K padding
__global__ void k_cvt(const float* __restrict__ src, bf16* __restrict__ dst,
                      int rows, int srcK, int dstK) {
  size_t total = (size_t)rows * dstK;
  for (size_t i = (size_t)blockIdx.x * blockDim.x + threadIdx.x; i < total;
       i += (size_t)gridDim.x * blockDim.x) {
    int r = (int)(i / dstK), k = (int)(i % dstK);
    dst[i] = (k < srcK) ? (bf16)src[(size_t)r * srcK + k] : (bf16)0.f;
  }
}

// xproj[2048,1536] = A[2048,Kp] @ W[1536,Kp]^T + bias  (one 16x64 tile per wave)
__global__ __launch_bounds__(256) void k_gemm_xproj(
    const bf16* __restrict__ A, const bf16* __restrict__ W,
    const float* __restrict__ bias, float* __restrict__ out, int Kp) {
  int gid = blockIdx.x * 8 + (threadIdx.x >> 5);
  int mt = gid / (G3H / 64);
  int ng = gid % (G3H / 64);
  int kt = Kp >> 5;
  v8f acc0 = {0,0,0,0,0,0,0,0}, acc1 = acc0, acc2 = acc0, acc3 = acc0;
  const bf16* Ab = A + (size_t)(mt * 16) * Kp;
  const bf16* Wb = W + (size_t)(ng * 64) * Kp;
  for (int k = 0; k < kt; ++k) {
    v16bf a = load_afrag(Ab + k * 32, Kp);
    acc0 = wmma_bf16(a, load_bfrag(Wb + (size_t)0  * Kp + k * 32, Kp), acc0);
    acc1 = wmma_bf16(a, load_bfrag(Wb + (size_t)16 * Kp + k * 32, Kp), acc1);
    acc2 = wmma_bf16(a, load_bfrag(Wb + (size_t)32 * Kp + k * 32, Kp), acc2);
    acc3 = wmma_bf16(a, load_bfrag(Wb + (size_t)48 * Kp + k * 32, Kp), acc3);
  }
  int lane = threadIdx.x & 31, n = lane & 15, hi = lane >> 4;
  v8f accs[4] = {acc0, acc1, acc2, acc3};
  #pragma unroll
  for (int st = 0; st < 4; ++st) {
    int col = ng * 64 + st * 16 + n;
    float bv = bias[col];
    #pragma unroll
    for (int j = 0; j < 8; ++j) {
      int r = mt * 16 + j + 8 * hi;
      out[(size_t)r * G3H + col] = accs[st][j] + bv;
    }
  }
}

// Persistent recurrent scan. 32 WGs x 96 threads (3 waves). WG `wg` owns
// h elements [16wg,16wg+16); its 48 w_hh rows (r,z,n) stay in LDS.
// One agent-scope spin barrier per time step.
__global__ __launch_bounds__(96) void k_scan(
    const float* __restrict__ xproj,      // [2048,1536], row = t*8+m
    const bf16*  __restrict__ whh,        // [1536,512] bf16
    const float* __restrict__ bhh,        // [1536]
    bf16*        __restrict__ outx,       // [2048,512]; row = t*mul_t + m*mul_m
    int mul_t, int mul_m,
    bf16*        __restrict__ hbuf,       // [2][8][512] bf16 ping-pong
    unsigned*    __restrict__ ctr)        // [256] zeroed counters
{
  __shared__ bf16 wsl[48 * 512];          // 48 KB
  __shared__ bf16 hbf[8 * 512];           // 8 KB  (h_prev, full H)
  __shared__ float hp[16 * 48];           // 3 KB
  __shared__ float bsl[48];
  const int wg  = blockIdx.x;             // 0..31
  const int tid = threadIdx.x;            // 0..95

  // stage w_hh slice: local row c = g*16+ii -> global row g*512 + wg*16 + ii
  for (int i = tid; i < 48 * 256; i += 96) {
    int c = i >> 8, k2 = i & 255;
    int g = c >> 4, ii = c & 15;
    int e = g * 512 + wg * 16 + ii;
    ((unsigned*)wsl)[c * 256 + k2] = ((const unsigned*)whh)[e * 256 + k2];
  }
  for (int c = tid; c < 48; c += 96) {
    int g = c >> 4, ii = c & 15;
    bsl[c] = bhh[g * 512 + wg * 16 + ii];
  }
  __syncthreads();

  const int wv = tid >> 5;
  const int lane = tid & 31;
  const int n = lane & 15, hi = lane >> 4;

  for (int t = 0; t < SS; ++t) {
    // fetch full h_prev (written by all WGs) with agent-scope loads
    unsigned* hsrc = (unsigned*)hbuf + (size_t)(t & 1) * 2048;
    for (int i = tid; i < 2048; i += 96)
      ((unsigned*)hbf)[i] = __hip_atomic_load(&hsrc[i], __ATOMIC_RELAXED, __HIP_MEMORY_SCOPE_AGENT);
    __syncthreads();

    // hp[m][c] = (h @ whh_slice^T)[m][c]; wave wv owns cols [16wv,16wv+16)
    v8f acc = {0,0,0,0,0,0,0,0};
    const bf16* wb = wsl + wv * 16 * 512;
    for (int k = 0; k < 16; ++k) {
      v16bf a = load_afrag_m8(hbf + k * 32, 512);
      v16bf b = load_bfrag(wb + k * 32, 512);
      acc = wmma_bf16(a, b, acc);
    }
    #pragma unroll
    for (int j = 0; j < 8; ++j)
      hp[(j + 8 * hi) * 48 + wv * 16 + n] = acc[j];
    __syncthreads();

    // gates: 64 items = 8 batch x 8 element-pairs
    if (tid < 64) {
      int m = tid >> 3, i2 = tid & 7;
      unsigned pk = 0;
      #pragma unroll
      for (int q = 0; q < 2; ++q) {
        int ii = i2 * 2 + q;            // 0..15 within WG
        int e  = wg * 16 + ii;          // 0..511 global h index
        size_t ro = (size_t)(t * 8 + m) * G3H;
        float xr = xproj[ro + e];
        float xz = xproj[ro + 512 + e];
        float xn = xproj[ro + 1024 + e];
        float hr = hp[m * 48 + ii]      + bsl[ii];
        float hz = hp[m * 48 + 16 + ii] + bsl[16 + ii];
        float hn = hp[m * 48 + 32 + ii] + bsl[32 + ii];
        float r  = 1.f / (1.f + __expf(-(xr + hr)));
        float z  = 1.f / (1.f + __expf(-(xz + hz)));
        float nn = tanhf(xn + r * hn);
        float hprev = (float)hbf[m * 512 + e];
        float hnew  = (1.f - z) * nn + z * hprev;
        bf16 hb = (bf16)hnew;
        unsigned short us = __builtin_bit_cast(unsigned short, hb);
        pk |= ((unsigned)us) << (16 * q);
      }
      int e0 = wg * 16 + i2 * 2;
      unsigned* hdst = (unsigned*)hbuf + (size_t)((t + 1) & 1) * 2048;
      __hip_atomic_store(&hdst[m * 256 + wg * 8 + i2], pk, __ATOMIC_RELAXED, __HIP_MEMORY_SCOPE_AGENT);
      size_t orow = (size_t)(t * mul_t + m * mul_m) * HH;
      *(unsigned*)(outx + orow + e0) = pk;
    }
    __syncthreads();
    if (tid == 0) {
      __hip_atomic_fetch_add(&ctr[t], 1u, __ATOMIC_ACQ_REL, __HIP_MEMORY_SCOPE_AGENT);
      while (__hip_atomic_load(&ctr[t], __ATOMIC_ACQUIRE, __HIP_MEMORY_SCOPE_AGENT) < (unsigned)SCAN_WGS) {
        __builtin_amdgcn_s_sleep(1);
      }
    }
    __syncthreads();
  }
}

// Z[2048,N] = A[2048,512] @ W[N,512]^T + bias. One block per 64-col group:
// stage W block to LDS as bf16 once (W read exactly once from HBM),
// then 8 waves sweep all 128 M-tiles.
__global__ __launch_bounds__(256) void k_gemm_out(
    const bf16* __restrict__ A, const float* __restrict__ W,
    const float* __restrict__ bias, float* __restrict__ Z, int N) {
  __shared__ bf16 wlds[64 * 512];       // 64 KB
  const int cb = blockIdx.x * 64;
  for (int i = threadIdx.x; i < 64 * 256; i += 256) {
    int r = i >> 8, k2 = i & 255;
    int wr = cb + r; if (wr >= N) wr = N - 1;
    const float* ps = W + (size_t)wr * HH + k2 * 2;
    wlds[r * 512 + k2 * 2]     = (bf16)ps[0];
    wlds[r * 512 + k2 * 2 + 1] = (bf16)ps[1];
  }
  __syncthreads();
  const int wv = threadIdx.x >> 5, lane = threadIdx.x & 31;
  const int n = lane & 15, hi = lane >> 4;
  for (int mt = wv; mt < 128; mt += 8) {
    v8f acc0 = {0,0,0,0,0,0,0,0}, acc1 = acc0, acc2 = acc0, acc3 = acc0;
    const bf16* Ab = A + (size_t)(mt * 16) * HH;
    for (int k = 0; k < 16; ++k) {
      v16bf a = load_afrag(Ab + k * 32, HH);
      acc0 = wmma_bf16(a, load_bfrag(wlds + 0  * 512 + k * 32, 512), acc0);
      acc1 = wmma_bf16(a, load_bfrag(wlds + 16 * 512 + k * 32, 512), acc1);
      acc2 = wmma_bf16(a, load_bfrag(wlds + 32 * 512 + k * 32, 512), acc2);
      acc3 = wmma_bf16(a, load_bfrag(wlds + 48 * 512 + k * 32, 512), acc3);
    }
    v8f accs[4] = {acc0, acc1, acc2, acc3};
    #pragma unroll
    for (int st = 0; st < 4; ++st) {
      int col = cb + st * 16 + n;
      if (col < N) {
        float bv = bias[col];
        #pragma unroll
        for (int j = 0; j < 8; ++j) {
          int r = mt * 16 + j + 8 * hi;
          Z[(size_t)r * N + col] = accs[st][j] + bv;
        }
      }
    }
  }
}

// per-row logsumexp (online), one block per row
__global__ __launch_bounds__(256) void k_rowstats(const float* __restrict__ Z, int N,
                                                  float* __restrict__ lse) {
  __shared__ float sm[256], ss[256];
  int r = blockIdx.x;
  const float* row = Z + (size_t)r * N;
  float m = -INFINITY, s = 0.f;
  for (int c = threadIdx.x; c < N; c += 256) {
    float v = row[c];
    float nm = fmaxf(m, v);
    s = s * __expf(m - nm) + __expf(v - nm);
    m = nm;
  }
  sm[threadIdx.x] = m; ss[threadIdx.x] = s;
  __syncthreads();
  for (int off = 128; off; off >>= 1) {
    if (threadIdx.x < off) {
      float m1 = sm[threadIdx.x], s1 = ss[threadIdx.x];
      float m2 = sm[threadIdx.x + off], s2 = ss[threadIdx.x + off];
      float nm = fmaxf(m1, m2);
      sm[threadIdx.x] = nm;
      ss[threadIdx.x] = s1 * __expf(m1 - nm) + s2 * __expf(m2 - nm);
    }
    __syncthreads();
  }
  if (threadIdx.x == 0) lse[r] = sm[0] + __logf(ss[0]);
}

__global__ void k_finalize(float* __restrict__ Z, int N, const float* __restrict__ lse) {
  size_t total4 = (size_t)ROWS * N / 4;
  for (size_t i = (size_t)blockIdx.x * blockDim.x + threadIdx.x; i < total4;
       i += (size_t)gridDim.x * blockDim.x) {
    int r = (int)((i * 4) / (size_t)N);
    float L = lse[r];
    float4 v = ((const float4*)Z)[i];
    v.x -= L; v.y -= L; v.z -= L; v.w -= L;
    ((float4*)Z)[i] = v;
  }
}

// ---------------- launch --------------------------------------------------
extern "C" void kernel_launch(void* const* d_in, const int* in_sizes, int n_in,
                              void* d_out, int out_size, void* d_ws, size_t ws_size,
                              hipStream_t stream) {
  const int*   word_idx = (const int*)d_in[0];
  const int*   sense_idx= (const int*)d_in[1];
  const float* X        = (const float*)d_in[2];
  const float* w_ih0    = (const float*)d_in[3];
  const float* w_hh0    = (const float*)d_in[4];
  const float* b_ih0    = (const float*)d_in[5];
  const float* b_hh0    = (const float*)d_in[6];
  const float* w_ih1    = (const float*)d_in[7];
  const float* w_hh1    = (const float*)d_in[8];
  const float* b_ih1    = (const float*)d_in[9];
  const float* b_hh1    = (const float*)d_in[10];
  const float* w_glob   = (const float*)d_in[11];
  const float* b_glob   = (const float*)d_in[12];
  const float* w_sen    = (const float*)d_in[13];
  const float* b_sen    = (const float*)d_in[14];

  char* ws = (char*)d_ws;
  size_t off = 0;
  auto alloc = [&](size_t bytes) -> void* {
    void* p = ws + off;
    off = (off + bytes + 255) & ~(size_t)255;
    return p;
  };
  bf16* x0    = (bf16*)alloc((size_t)ROWS * KP0 * 2);
  bf16* wih0b = (bf16*)alloc((size_t)G3H * KP0 * 2);
  bf16* whh0b = (bf16*)alloc((size_t)G3H * HH * 2);
  bf16* wih1b = (bf16*)alloc((size_t)G3H * HH * 2);
  bf16* whh1b = (bf16*)alloc((size_t)G3H * HH * 2);
  float* xproj= (float*)alloc((size_t)ROWS * G3H * 4);   // reused by both layers
  bf16* hx0   = (bf16*)alloc((size_t)ROWS * HH * 2);
  bf16* gru   = (bf16*)alloc((size_t)ROWS * HH * 2);
  bf16* hb0   = (bf16*)alloc((size_t)2 * 8 * HH * 2);
  bf16* hb1   = (bf16*)alloc((size_t)2 * 8 * HH * 2);
  unsigned* ctr = (unsigned*)alloc(512 * 4);
  float* lseg = (float*)alloc(ROWS * 4);
  float* lses = (float*)alloc(ROWS * 4);

  k_init<<<8, 256, 0, stream>>>(ctr, hb0, hb1);
  k_embed<<<ROWS, 128, 0, stream>>>(word_idx, sense_idx, X, x0);
  k_cvt<<<512, 256, 0, stream>>>(w_ih0, wih0b, G3H, 600, KP0);
  k_cvt<<<512, 256, 0, stream>>>(w_hh0, whh0b, G3H, HH, HH);
  k_cvt<<<512, 256, 0, stream>>>(w_ih1, wih1b, G3H, HH, HH);
  k_cvt<<<512, 256, 0, stream>>>(w_hh1, whh1b, G3H, HH, HH);

  // layer 0
  k_gemm_xproj<<<(128 * (G3H / 64)) / 8, 256, 0, stream>>>(x0, wih0b, b_ih0, xproj, KP0);
  k_scan<<<SCAN_WGS, 96, 0, stream>>>(xproj, whh0b, b_hh0, hx0, /*mul_t*/8, /*mul_m*/1, hb0, ctr);
  // layer 1 (output written directly in b*S+s order)
  k_gemm_xproj<<<(128 * (G3H / 64)) / 8, 256, 0, stream>>>(hx0, wih1b, b_ih1, xproj, HH);
  k_scan<<<SCAN_WGS, 96, 0, stream>>>(xproj, whh1b, b_hh1, gru, /*mul_t*/1, /*mul_m*/SS, hb1, ctr + 256);

  float* zg = (float*)d_out;
  float* zs = zg + (size_t)ROWS * NGL;
  k_gemm_out<<<(NGL + 63) / 64, 256, 0, stream>>>(gru, w_glob, b_glob, zg, NGL);
  k_gemm_out<<<(NSE + 63) / 64, 256, 0, stream>>>(gru, w_sen, b_sen, zs, NSE);

  k_rowstats<<<ROWS, 256, 0, stream>>>(zg, NGL, lseg);
  k_rowstats<<<ROWS, 256, 0, stream>>>(zs, NSE, lses);
  k_finalize<<<4096, 256, 0, stream>>>(zg, NGL, lseg);
  k_finalize<<<2048, 256, 0, stream>>>(zs, NSE, lses);
}
